// DecoderWithoutAttention_37022618091836
// MI455X (gfx1250) — compile-verified, matrix operand order checked by reading, not run on hardware
//
#include <hip/hip_runtime.h>

// Problem constants
#define BB   128
#define PP   8
#define DENC 1280
#define LL   50
#define VV   10000
#define EE   512
#define HH   512
#define TT   (LL - 1)        // 49 decode steps
#define G4   (4 * HH)        // 2048 gates
#define KIH  (EE + DENC)     // 1792 = W_ih inner dim

typedef __bf16        v16bf __attribute__((ext_vector_type(16)));
typedef float         v8f   __attribute__((ext_vector_type(8)));
typedef unsigned int  v8u   __attribute__((ext_vector_type(8)));

__device__ __forceinline__ unsigned short f2bf(float f) {
  unsigned int u = __builtin_bit_cast(unsigned int, f);
  u += 0x7fffu + ((u >> 16) & 1u);            // round-to-nearest-even
  return (unsigned short)(u >> 16);
}
__device__ __forceinline__ float sigf(float x) { return 1.0f / (1.0f + __expf(-x)); }

// Fragment loader: 16 bf16 (as 8 dwords) following ISA 7.12.2 16-bit A/B layout.
// row = lane&15 selects the matrix row (A) / weight row = B column (B^T load),
// half = lane>>4 selects the K sub-block.
__device__ __forceinline__ v16bf load_frag(const unsigned short* __restrict__ rowp,
                                           int kc, int half) {
  v8u u;
#pragma unroll
  for (int v = 0; v < 8; ++v) {
    const int k0 = kc + ((v & 4) << 2) + (half << 3) + ((v & 3) << 1);
    u[v] = *(const unsigned int*)(rowp + k0);
  }
  return __builtin_bit_cast(v16bf, u);
}

// ---------------------------------------------------------------------------
// 1) Mean-pool encoder_out (B,8,8,1280) -> enc_raw (B,1280)
__global__ void enc_mean(const float* __restrict__ eo, float* __restrict__ enc_raw) {
  int tid = blockIdx.x * blockDim.x + threadIdx.x;
  if (tid >= BB * DENC) return;
  int b = tid / DENC, d = tid % DENC;
  const float* p = eo + (size_t)b * (PP * PP) * DENC + d;
  float s = 0.f;
#pragma unroll 4
  for (int q = 0; q < PP * PP; ++q) s += p[(size_t)q * DENC];
  enc_raw[tid] = s * (1.0f / (PP * PP));
}

// 2) Stable descending rank-sort by length (B=128, single block), emit the
//    integer outputs (caps sorted, decode_lengths, sort_ind) as floats.
__global__ void sort_and_emit(const int* __restrict__ caps,
                              const int* __restrict__ clen,
                              int* __restrict__ sort_ind,
                              int* __restrict__ dlen,
                              int* __restrict__ caps_sorted,
                              float* __restrict__ d_out) {
  __shared__ int lens[BB];
  const int i = threadIdx.x;
  lens[i] = clen[i];                     // (B,1)
  __syncthreads();
  const int li = lens[i];
  int r = 0;
  for (int j = 0; j < BB; ++j) {
    int lj = lens[j];
    if (lj > li || (lj == li && j < i)) ++r;   // stable argsort(-lens)
  }
  sort_ind[r] = i;
  dlen[r] = li - 1;
  const size_t predN = (size_t)BB * TT * VV;
  for (int tt = 0; tt < LL; ++tt) {
    int cv = caps[i * LL + tt];
    caps_sorted[r * LL + tt] = cv;
    d_out[predN + (size_t)r * LL + tt] = (float)cv;
  }
  d_out[predN + (size_t)BB * LL + r]      = (float)(li - 1);   // decode_lengths
  d_out[predN + (size_t)BB * LL + BB + r] = (float)i;          // sort_ind
}

// 3) Gather enc rows in sorted order and convert to bf16
__global__ void gather_enc(const float* __restrict__ enc_raw,
                           const int* __restrict__ sort_ind,
                           unsigned short* __restrict__ enc_bf) {
  int tid = blockIdx.x * blockDim.x + threadIdx.x;
  if (tid >= BB * DENC) return;
  int r = tid / DENC, d = tid % DENC;
  enc_bf[tid] = f2bf(enc_raw[(size_t)sort_ind[r] * DENC + d]);
}

// 4) Generic f32 -> bf16 conversion
__global__ void cvt_bf16(const float* __restrict__ src, unsigned short* __restrict__ dst, int n) {
  int tid = blockIdx.x * blockDim.x + threadIdx.x;
  if (tid < n) dst[tid] = f2bf(src[tid]);
}

// 5) Embedding lookup (sorted caps) -> bf16, layout (B, T, E)
__global__ void emb_gather(const float* __restrict__ embW,
                           const int* __restrict__ caps_sorted,
                           unsigned short* __restrict__ out) {
  int tid = blockIdx.x * blockDim.x + threadIdx.x;
  if (tid >= BB * TT * EE) return;
  int e = tid % EE;
  int bt = tid / EE;
  int t = bt % TT, b = bt / TT;
  int tok = caps_sorted[b * LL + t];
  out[tid] = f2bf(embW[(size_t)tok * EE + e]);
}

// ---------------------------------------------------------------------------
// WMMA GEMM, one wave computes an MT-tall column of 16x16 tiles (MT*16 x 16),
// sharing each B (weight) fragment across MT wmmas. Optionally accumulates a
// second GEMM (A2 x W2^T) into the same accumulators (fused gate GEMMs):
//   out(MxN) = A1(M x K1) @ W1(N x K1)^T [+ A2(M x K2) @ W2(N x K2)^T]
//              + bias + bias2 + addsrc ;  optional bf16 mirror of out.
template <int MT>
__global__ void wmma_gemm2(const unsigned short* __restrict__ A1, int lda1,
                           const unsigned short* __restrict__ W1, int ldw1, int K1,
                           const unsigned short* __restrict__ A2, int lda2,
                           const unsigned short* __restrict__ W2, int ldw2, int K2,
                           const float* __restrict__ bias,
                           const float* __restrict__ bias2,
                           const float* __restrict__ addsrc,
                           float* __restrict__ out,
                           unsigned short* __restrict__ out_bf,
                           int M, int N) {
  const int wave = (blockIdx.x * blockDim.x + threadIdx.x) >> 5;
  const int lane = threadIdx.x & 31;
  const int tilesN  = N >> 4;
  const int mGroups = M / (16 * MT);
  if (wave >= mGroups * tilesN) return;
  const int tn = wave % tilesN;
  const int mg = wave / tilesN;
  const int col  = lane & 15;
  const int half = lane >> 4;

  v8f acc[MT];
#pragma unroll
  for (int i = 0; i < MT; ++i) acc[i] = (v8f){};

  // Pass 1
  {
    const unsigned short* Wrow = W1 + (size_t)(tn * 16 + col) * ldw1;
    const unsigned short* Abase = A1 + (size_t)(mg * MT * 16 + col) * lda1;
    for (int kc = 0; kc < K1; kc += 32) {
      v16bf bf = load_frag(Wrow, kc, half);
#pragma unroll
      for (int i = 0; i < MT; ++i) {
        v16bf af = load_frag(Abase + (size_t)(i * 16) * lda1, kc, half);
        acc[i] = __builtin_amdgcn_wmma_f32_16x16x32_bf16(
            false, af, false, bf, (short)0, acc[i], false, false);
      }
    }
  }
  // Optional fused second GEMM
  if (A2) {
    const unsigned short* Wrow = W2 + (size_t)(tn * 16 + col) * ldw2;
    const unsigned short* Abase = A2 + (size_t)(mg * MT * 16 + col) * lda2;
    for (int kc = 0; kc < K2; kc += 32) {
      v16bf bf = load_frag(Wrow, kc, half);
#pragma unroll
      for (int i = 0; i < MT; ++i) {
        v16bf af = load_frag(Abase + (size_t)(i * 16) * lda2, kc, half);
        acc[i] = __builtin_amdgcn_wmma_f32_16x16x32_bf16(
            false, af, false, bf, (short)0, acc[i], false, false);
      }
    }
  }

  const int n = tn * 16 + col;
  float badd = 0.f;
  if (bias)  badd += bias[n];
  if (bias2) badd += bias2[n];
#pragma unroll
  for (int i = 0; i < MT; ++i) {
#pragma unroll
    for (int v = 0; v < 8; ++v) {
      const int m = mg * MT * 16 + i * 16 + v + (half << 3);
      float r = acc[i][v] + badd;
      if (addsrc) r += addsrc[(size_t)m * N + n];
      out[(size_t)m * N + n] = r;
      if (out_bf) out_bf[(size_t)m * N + n] = f2bf(r);
    }
  }
}

// 6) LSTM elementwise: gates(B,2048) -> h,c update with activity mask
__global__ void lstm_elem(const float* __restrict__ gates,
                          float* __restrict__ h, float* __restrict__ c,
                          unsigned short* __restrict__ h_bf,
                          unsigned short* __restrict__ hn_bf,
                          const int* __restrict__ dlen, int t) {
  int tid = blockIdx.x * blockDim.x + threadIdx.x;
  if (tid >= BB * HH) return;
  int b = tid / HH, j = tid % HH;
  const float* g = gates + (size_t)b * G4;
  float ig = g[j], fg = g[HH + j], gg = g[2 * HH + j], og = g[3 * HH + j];
  float co = c[tid];
  float cn = sigf(fg) * co + sigf(ig) * tanhf(gg);
  float hn = sigf(og) * tanhf(cn);
  bool act = t < dlen[b];
  float h2 = act ? hn : h[tid];
  float c2 = act ? cn : co;
  h[tid] = h2;
  c[tid] = c2;
  h_bf[tid]  = f2bf(h2);   // recurrent input (masked)
  hn_bf[tid] = f2bf(hn);   // logits input (unmasked, per reference)
}

// 7) fc logits: preds(b,t,:) = h_new @ fc_W^T + fc_b, zeroed where inactive.
//    MT-tall wave tiles share each fc_W fragment across MT wmmas.
template <int MT>
__global__ void wmma_fc(const unsigned short* __restrict__ Hn,
                        const unsigned short* __restrict__ Wf,
                        const float* __restrict__ fb,
                        const int* __restrict__ dlen,
                        int t, float* __restrict__ out) {
  const int wave = (blockIdx.x * blockDim.x + threadIdx.x) >> 5;
  const int lane = threadIdx.x & 31;
  const int tilesN  = VV >> 4;          // 625
  const int mGroups = BB / (16 * MT);
  if (wave >= mGroups * tilesN) return;
  const int tn = wave % tilesN;
  const int mg = wave / tilesN;
  const int col = lane & 15, half = lane >> 4;

  v8f acc[MT];
#pragma unroll
  for (int i = 0; i < MT; ++i) acc[i] = (v8f){};

  const unsigned short* Wrow = Wf + (size_t)(tn * 16 + col) * HH;
  const unsigned short* Abase = Hn + (size_t)(mg * MT * 16 + col) * HH;
  for (int kc = 0; kc < HH; kc += 32) {
    v16bf bf = load_frag(Wrow, kc, half);
#pragma unroll
    for (int i = 0; i < MT; ++i) {
      v16bf af = load_frag(Abase + (size_t)(i * 16) * HH, kc, half);
      acc[i] = __builtin_amdgcn_wmma_f32_16x16x32_bf16(
          false, af, false, bf, (short)0, acc[i], false, false);
    }
  }

  const int n = tn * 16 + col;
  const float bn = fb[n];
#pragma unroll
  for (int i = 0; i < MT; ++i) {
#pragma unroll
    for (int v = 0; v < 8; ++v) {
      const int b = mg * MT * 16 + i * 16 + v + (half << 3);
      const bool act = t < dlen[b];
      out[(size_t)b * ((size_t)TT * VV) + (size_t)t * VV + n] =
          act ? (acc[i][v] + bn) : 0.0f;
    }
  }
}

// ---------------------------------------------------------------------------
extern "C" void kernel_launch(void* const* d_in, const int* in_sizes, int n_in,
                              void* d_out, int out_size, void* d_ws, size_t ws_size,
                              hipStream_t stream) {
  const float* encoder_out = (const float*)d_in[0];
  const int*   caps        = (const int*)d_in[1];
  const int*   clen        = (const int*)d_in[2];
  const float* embW        = (const float*)d_in[3];
  const float* W_ih        = (const float*)d_in[4];
  const float* W_hh        = (const float*)d_in[5];
  const float* b_ih        = (const float*)d_in[6];
  const float* b_hh        = (const float*)d_in[7];
  const float* fc_W        = (const float*)d_in[8];
  const float* fc_b        = (const float*)d_in[9];
  const float* ihW         = (const float*)d_in[10];
  const float* ihb         = (const float*)d_in[11];
  const float* icW         = (const float*)d_in[12];
  const float* icb         = (const float*)d_in[13];
  float* out = (float*)d_out;

  // Workspace carve-up (256B aligned)
  char* ws = (char*)d_ws;
  size_t off = 0;
  auto carve = [&](size_t bytes) -> char* {
    char* p = ws + off;
    off += (bytes + 255) & ~(size_t)255;
    return p;
  };
  float*          enc_raw     = (float*)carve((size_t)BB * DENC * 4);
  unsigned short* enc_bf      = (unsigned short*)carve((size_t)BB * DENC * 2);
  int*            sort_ind    = (int*)carve(BB * 4);
  int*            dlen        = (int*)carve(BB * 4);
  int*            caps_sorted = (int*)carve((size_t)BB * LL * 4);
  unsigned short* Wih_bf      = (unsigned short*)carve((size_t)G4 * KIH * 2);
  unsigned short* Whh_bf      = (unsigned short*)carve((size_t)G4 * HH * 2);
  unsigned short* fcW_bf      = (unsigned short*)carve((size_t)VV * HH * 2);
  unsigned short* ihW_bf      = (unsigned short*)carve((size_t)HH * DENC * 2);
  unsigned short* icW_bf      = (unsigned short*)carve((size_t)HH * DENC * 2);
  unsigned short* emb_bf      = (unsigned short*)carve((size_t)BB * TT * EE * 2);
  float*          gates_enc   = (float*)carve((size_t)BB * G4 * 4);
  float*          gates       = (float*)carve((size_t)BB * G4 * 4);
  float*          h_f         = (float*)carve((size_t)BB * HH * 4);
  float*          c_f         = (float*)carve((size_t)BB * HH * 4);
  unsigned short* h_bf        = (unsigned short*)carve((size_t)BB * HH * 2);
  unsigned short* hn_bf       = (unsigned short*)carve((size_t)BB * HH * 2);

  auto nb = [](long n) { return (int)((n + 255) / 256); };
  auto gemmBlocks = [](int M, int N, int MT) {
    long waves = (long)(M / (16 * MT)) * (N / 16);
    return (int)((waves + 7) / 8);          // 8 waves per 256-thread block
  };

  // Setup
  enc_mean<<<nb((long)BB * DENC), 256, 0, stream>>>(encoder_out, enc_raw);
  sort_and_emit<<<1, BB, 0, stream>>>(caps, clen, sort_ind, dlen, caps_sorted, out);
  gather_enc<<<nb((long)BB * DENC), 256, 0, stream>>>(enc_raw, sort_ind, enc_bf);
  cvt_bf16<<<nb((long)G4 * KIH), 256, 0, stream>>>(W_ih, Wih_bf, G4 * KIH);
  cvt_bf16<<<nb((long)G4 * HH), 256, 0, stream>>>(W_hh, Whh_bf, G4 * HH);
  cvt_bf16<<<nb((long)VV * HH), 256, 0, stream>>>(fc_W, fcW_bf, VV * HH);
  cvt_bf16<<<nb((long)HH * DENC), 256, 0, stream>>>(ihW, ihW_bf, HH * DENC);
  cvt_bf16<<<nb((long)HH * DENC), 256, 0, stream>>>(icW, icW_bf, HH * DENC);
  emb_gather<<<nb((long)BB * TT * EE), 256, 0, stream>>>(embW, caps_sorted, emb_bf);

  // h0 = enc @ init_h_W^T + b ; c0 = enc @ init_c_W^T + b   (K = 1280)
  wmma_gemm2<2><<<gemmBlocks(BB, HH, 2), 256, 0, stream>>>(
      enc_bf, DENC, ihW_bf, DENC, DENC,
      nullptr, 0, nullptr, 0, 0,
      ihb, nullptr, nullptr, h_f, h_bf, BB, HH);
  wmma_gemm2<2><<<gemmBlocks(BB, HH, 2), 256, 0, stream>>>(
      enc_bf, DENC, icW_bf, DENC, DENC,
      nullptr, 0, nullptr, 0, 0,
      icb, nullptr, nullptr, c_f, nullptr, BB, HH);
  // Time-invariant gate term: enc @ W_ih[:,512:]^T + b_ih + b_hh  (K = 1280)
  wmma_gemm2<2><<<gemmBlocks(BB, G4, 2), 256, 0, stream>>>(
      enc_bf, DENC, Wih_bf + EE, KIH, DENC,
      nullptr, 0, nullptr, 0, 0,
      b_ih, b_hh, nullptr, gates_enc, nullptr, BB, G4);

  // Sequential decode: fused gate GEMM -> LSTM elementwise -> masked fc logits
  for (int t = 0; t < TT; ++t) {
    // gates = emb_t @ W_ih[:, :512]^T + h @ W_hh^T + gates_enc
    wmma_gemm2<2><<<gemmBlocks(BB, G4, 2), 256, 0, stream>>>(
        emb_bf + (size_t)t * EE, TT * EE, Wih_bf, KIH, EE,
        h_bf, HH, Whh_bf, HH, HH,
        nullptr, nullptr, gates_enc, gates, nullptr, BB, G4);
    lstm_elem<<<nb((long)BB * HH), 256, 0, stream>>>(gates, h_f, c_f, h_bf, hn_bf, dlen, t);
    // preds[:, t, :] = h_new @ fc_W^T + fc_b (masked)
    wmma_fc<4><<<gemmBlocks(BB, VV, 4), 256, 0, stream>>>(hn_bf, fcW_bf, fc_b, dlen, t, out);
  }
}
